// MultiHeadAggregationNetwork_47528108098314
// MI455X (gfx1250) — matrix operand order repaired and checked by reading, not compiled, not run on hardware
//
#include <hip/hip_runtime.h>
#include <hip/hip_bf16.h>

typedef __attribute__((ext_vector_type(16))) __bf16 v16bf;
typedef __attribute__((ext_vector_type(8)))  float  v8f;
typedef __attribute__((ext_vector_type(4)))  unsigned int u32x4;
typedef __attribute__((ext_vector_type(8)))  int i32x8;
typedef __attribute__((ext_vector_type(4)))  int i32x4;

#define N_ROWS 131072   // B*O
#define B_     32
#define O_     4096
#define F_     256
#define HID_   128
#define H_     8

__device__ __forceinline__ v8f zero8() {
  v8f z;
#pragma unroll
  for (int i = 0; i < 8; ++i) z[i] = 0.f;
  return z;
}

// Load one pre-swizzled 512-element (32 lanes x 16) WMMA operand tile:
// each lane reads 32 contiguous bytes -> 2x b128 loads, no repacking.
__device__ __forceinline__ v16bf load_tile16(const __bf16* p, int tile, int lane) {
  return *(const v16bf*)(p + (((tile << 5) + lane) << 4));
}

// A-operand swizzle helpers (16x32 bf16 tile):
// lane = 16*(k>>4) + m ; elem = k&15
__device__ __forceinline__ int a_lane(int kk, int m) { return ((kk >> 4) << 4) + m; }
__device__ __forceinline__ int a_elem(int kk) { return kk & 15; }

// ---------------- K0: weight conversion into WMMA-operand order ----------------
// B-tile element mapping: value[e] at lane L = W[k0 + 16*(L>>4) + 2*(e>>1) + (e&1)][n0 + (L&15)]
__global__ void k0_prep(const float* __restrict__ Wi, const float* __restrict__ Wb,
                        const float* __restrict__ Wf,
                        __bf16* __restrict__ Wi_sw, __bf16* __restrict__ Wb_sw,
                        __bf16* __restrict__ Wf_sw, float* __restrict__ stats) {
  int t = blockIdx.x * blockDim.x + threadIdx.x;
  int stride = gridDim.x * blockDim.x;
  // W_in: K=256 (8 kt) x N=128 (8 nt) -> 64 tiles
  for (int i = t; i < 64 * 512; i += stride) {
    int tl = i >> 9, lane = (i >> 4) & 31, e = i & 15;
    int k = (tl >> 3) * 32 + ((lane >> 4) << 4) + 2 * (e >> 1) + (e & 1);
    int n = (tl & 7) * 16 + (lane & 15);
    Wi_sw[i] = (__bf16)Wi[k * HID_ + n];
  }
  // W_blk: K=128 (4 kt) x N=128 (8 nt) -> 32 tiles
  for (int i = t; i < 32 * 512; i += stride) {
    int tl = i >> 9, lane = (i >> 4) & 31, e = i & 15;
    int k = (tl >> 3) * 32 + ((lane >> 4) << 4) + 2 * (e >> 1) + (e & 1);
    int n = (tl & 7) * 16 + (lane & 15);
    Wb_sw[i] = (__bf16)Wb[k * HID_ + n];
  }
  // W_fc: K=128 (4 kt) x N=16 (pad of 8) -> 4 tiles
  for (int i = t; i < 4 * 512; i += stride) {
    int tl = i >> 9, lane = (i >> 4) & 31, e = i & 15;
    int k = tl * 32 + ((lane >> 4) << 4) + 2 * (e >> 1) + (e & 1);
    int n = lane & 15;
    Wf_sw[i] = (__bf16)((n < H_) ? Wf[k * H_ + n] : 0.f);
  }
  for (int i = t; i < 512; i += stride) stats[i] = 0.f;  // sum[128], ss[128], denom[256]
}

// ---------------- K1: GEMM1 + relu, GEMM2, BN stat accumulation ----------------
__global__ __launch_bounds__(128) void k1_gemm12(
    const float* __restrict__ x, const __bf16* __restrict__ Wi_sw,
    const float* __restrict__ b_in, const __bf16* __restrict__ Wb_sw,
    const float* __restrict__ b_blk, __bf16* __restrict__ h_out,
    float* __restrict__ z_out, float* __restrict__ mask_out,
    float* __restrict__ chan_sum, float* __restrict__ chan_ss) {
  __shared__ alignas(32) __bf16 sxA[4 * 8 * 512];  // 32 KB: A tiles (wave, kt)
  __shared__ alignas(32) __bf16 shA[4 * 4 * 512];  // 16 KB: A tiles for GEMM2
  __shared__ float srow[64];
  __shared__ float csum[128];
  __shared__ float css[128];

  int tid = threadIdx.x, lane = tid & 31, wave = tid >> 5;
  long long row0 = (long long)blockIdx.x * 64;

  if (tid < 64) srow[tid] = 0.f;
  csum[tid] = 0.f;
  css[tid]  = 0.f;
  __syncthreads();

  // stage x tile (float4 reads) into A-operand order + row sumsq for the mask
  const float4* x4 = reinterpret_cast<const float4*>(x + row0 * F_);
  for (int i4 = tid; i4 < 64 * 256 / 4; i4 += 128) {
    float4 vv = x4[i4];
    int i = i4 << 2;
    int r = i >> 8, c = i & 255;
    atomicAdd(&srow[r], vv.x * vv.x + vv.y * vv.y + vv.z * vv.z + vv.w * vv.w);
    int wv = r >> 4, m = r & 15;
    int kt = c >> 5, kk = c & 31;
    __bf16* dst = &sxA[((((wv << 3) + kt) << 5) + a_lane(kk, m)) * 16 + a_elem(kk)];
    dst[0] = (__bf16)vv.x; dst[1] = (__bf16)vv.y;
    dst[2] = (__bf16)vv.z; dst[3] = (__bf16)vv.w;
  }
  __syncthreads();
  if (tid < 64) mask_out[row0 + tid] = (srow[tid] != 0.f) ? 1.f : 0.f;

  int n_lo = lane & 15, half = lane >> 4;

  // GEMM1: h = relu(x @ W_in + b_in)   M=16 K=256 N=128
  v8f acc1[8];
#pragma unroll
  for (int nt = 0; nt < 8; ++nt) acc1[nt] = zero8();
#pragma unroll
  for (int kt = 0; kt < 8; ++kt) {
    v16bf a = load_tile16(sxA, (wave << 3) + kt, lane);
#pragma unroll
    for (int nt = 0; nt < 8; ++nt) {
      v16bf b = load_tile16(Wi_sw, (kt << 3) + nt, lane);
      acc1[nt] = __builtin_amdgcn_wmma_f32_16x16x32_bf16(
          false, a, false, b, (short)0, acc1[nt], false, false);
    }
  }
#pragma unroll
  for (int nt = 0; nt < 8; ++nt) {
    int n = nt * 16 + n_lo;
    float bi = b_in[n];
    int kt2 = n >> 5, kk = n & 31, e = a_elem(kk);
#pragma unroll
    for (int r = 0; r < 8; ++r) {
      int m = r + 8 * half;
      float hv = acc1[nt][r] + bi;
      hv = hv > 0.f ? hv : 0.f;
      __bf16 hb = (__bf16)hv;
      shA[((((wave << 2) + kt2) << 5) + a_lane(kk, m)) * 16 + e] = hb;
      h_out[(row0 + wave * 16 + m) * HID_ + n] = hb;  // row-major for K3
    }
  }
  __syncthreads();

  // GEMM2: z = h @ W_blk + b_blk   M=16 K=128 N=128
  v8f acc2[8];
#pragma unroll
  for (int nt = 0; nt < 8; ++nt) acc2[nt] = zero8();
#pragma unroll
  for (int kt = 0; kt < 4; ++kt) {
    v16bf a = load_tile16(shA, (wave << 2) + kt, lane);
#pragma unroll
    for (int nt = 0; nt < 8; ++nt) {
      v16bf b = load_tile16(Wb_sw, (kt << 3) + nt, lane);
      acc2[nt] = __builtin_amdgcn_wmma_f32_16x16x32_bf16(
          false, a, false, b, (short)0, acc2[nt], false, false);
    }
  }
#pragma unroll
  for (int nt = 0; nt < 8; ++nt) {
    int n = nt * 16 + n_lo;
    float bb = b_blk[n];
    float s = 0.f, ss = 0.f;
#pragma unroll
    for (int r = 0; r < 8; ++r) {
      int m = r + 8 * half;
      float zv = acc2[nt][r] + bb;
      z_out[(row0 + wave * 16 + m) * HID_ + n] = zv;
      s += zv;
      ss += zv * zv;
    }
    atomicAdd(&csum[n], s);
    atomicAdd(&css[n], ss);
  }
  __syncthreads();
  atomicAdd(&chan_sum[tid], csum[tid]);
  atomicAdd(&chan_ss[tid], css[tid]);
}

// ---------------- K2: finalize BN scale/shift ----------------
__global__ void k2_stats(const float* __restrict__ chan_sum, const float* __restrict__ chan_ss,
                         const float* __restrict__ gamma, const float* __restrict__ beta,
                         float* __restrict__ scale, float* __restrict__ shift) {
  int c = threadIdx.x;  // 128 threads
  const float invN = 1.f / (float)N_ROWS;
  float mu  = chan_sum[c] * invN;
  float var = chan_ss[c] * invN - mu * mu;
  float sc  = gamma[c] * rsqrtf(var + 1e-5f);
  scale[c] = sc;
  shift[c] = beta[c] - mu * sc;
}

// ---------------- K3: residual + fc + mask + gumbel exp + denom ----------------
__global__ __launch_bounds__(128) void k3_fc_exp(
    const float* __restrict__ z, const __bf16* __restrict__ h_bf,
    const __bf16* __restrict__ Wf_sw, const float* __restrict__ b_fc,
    const float* __restrict__ gum, const float* __restrict__ mask,
    const float* __restrict__ scale, const float* __restrict__ shift,
    float* __restrict__ e_out, float* __restrict__ denom) {
  __shared__ float ssc[128];
  __shared__ float ssh[128];
  __shared__ float sden[8];

  int tid = threadIdx.x, lane = tid & 31, wave = tid >> 5;
  long long row0 = (long long)blockIdx.x * 64 + wave * 16;

  ssc[tid] = scale[tid];
  ssh[tid] = shift[tid];
  if (tid < 8) sden[tid] = 0.f;
  __syncthreads();

  int m = lane & 15, halfk = lane >> 4;
  long long ridx = row0 + m;

  v8f acc = zero8();
#pragma unroll
  for (int kt = 0; kt < 4; ++kt) {
    v16bf a;
#pragma unroll
    for (int v = 0; v < 8; ++v) {
      int k = kt * 32 + halfk * 16 + (v << 1);
#pragma unroll
      for (int j = 0; j < 2; ++j) {
        int kk = k + j;
        float zv = z[ridx * HID_ + kk] * ssc[kk] + ssh[kk];
        zv = zv > 0.f ? zv : 0.f;
        a[2 * v + j] = (__bf16)(zv + (float)h_bf[ridx * HID_ + kk]);
      }
    }
    v16bf b = load_tile16(Wf_sw, kt, lane);
    acc = __builtin_amdgcn_wmma_f32_16x16x32_bf16(
        false, a, false, b, (short)0, acc, false, false);
  }

  int n = lane & 15, half = lane >> 4;
  if (n < H_) {
    float bfc = b_fc[n];
#pragma unroll
    for (int r = 0; r < 8; ++r) {
      long long row = row0 + r + 8 * half;
      float lg = (acc[r] + bfc) * mask[row];
      float ev = __expf(lg + gum[row * H_ + n]);  // TAU = 1
      e_out[row * H_ + n] = ev;
      atomicAdd(&sden[n], ev);
    }
  }
  __syncthreads();
  if (tid < 8) atomicAdd(&denom[(blockIdx.x >> 6) * H_ + tid], sden[tid]);
}

// ---------------- K4: pooled[b,h,f] = (sum_o e*x)/denom ----------------
__global__ __launch_bounds__(128) void k4_pool(
    const float* __restrict__ x, const float* __restrict__ e,
    const float* __restrict__ denom, float* __restrict__ out) {
  __shared__ alignas(32) float  se[256 * 8];          //  8 KB : e chunk (raw f32, TDM target)
  __shared__ alignas(32) __bf16 sxB[8 * 4 * 512];     // 32 KB : B tiles (ks, ntile)

  int tid = threadIdx.x, lane = tid & 31, wave = tid >> 5;
  int b = blockIdx.x;
  int f0 = blockIdx.y * 64;

  v8f acc = zero8();
  for (int k0 = 0; k0 < O_; k0 += 256) {
    __syncthreads();

    // ---- e chunk: 2048 contiguous f32 (8 KB) global -> LDS via Tensor Data Mover ----
#if __has_builtin(__builtin_amdgcn_tensor_load_to_lds)
    if (tid < 32) {  // one wave issues the TDM op (EXEC is ignored by TDM)
      unsigned lds_off = (unsigned)(unsigned long long)(&se[0]);
      unsigned long long ga =
          (unsigned long long)(&e[((long long)b * O_ + k0) * H_]);
      u32x4 g0;
      g0[0] = 1u;                                   // count=1, user descriptor
      g0[1] = lds_off;                              // lds_addr
      g0[2] = (unsigned)(ga & 0xffffffffu);         // global_addr[31:0]
      g0[3] = (unsigned)((ga >> 32) & 0x1ffffffu) | (2u << 30);  // addr[56:32] | type=2
      i32x8 g1;
      g1[0] = (int)(2u << 16);      // workgroup_mask=0 | data_size=2 (4B)
      g1[1] = (int)(2048u << 16);   // atomic_barrier_addr=0 | tensor_dim0[15:0]=2048
      g1[2] = (int)(1u << 16);      // tensor_dim0[31:16]=0 | tensor_dim1[15:0]=1
      g1[3] = (int)(2048u << 16);   // tensor_dim1[31:16]=0 | tile_dim0=2048
      g1[4] = 1;                    // tile_dim1=1, tile_dim2=0
      g1[5] = 2048;                 // tensor_dim0_stride[31:0]
      g1[6] = 0;
      g1[7] = 0;
      i32x4 gz = {0, 0, 0, 0};
#if __clang_major__ >= 23
      i32x8 gz8 = {0, 0, 0, 0, 0, 0, 0, 0};
      __builtin_amdgcn_tensor_load_to_lds(g0, g1, gz, gz, gz8, 0);
#else
      __builtin_amdgcn_tensor_load_to_lds(g0, g1, gz, gz, 0);
#endif
    }
    __builtin_amdgcn_s_wait_tensorcnt(0);
#else
    for (int i = tid; i < 256 * 8; i += 128)
      se[i] = e[((long long)b * O_ + k0) * H_ + i];
#endif

    // ---- x chunk: coalesced float4 reads, store into B-operand order ----
    for (int i4 = tid; i4 < 256 * 64 / 4; i4 += 128) {
      int i = i4 << 2;
      int kr = i >> 6, c = i & 63;
      const float4* xr = reinterpret_cast<const float4*>(
          &x[((long long)b * O_ + k0 + kr) * F_ + f0]);
      float4 vv = xr[c >> 2];
      if (k0 + 256 < O_) __builtin_prefetch((const float*)xr + 256 * F_, 0, 1);
      int ks = kr >> 5, kk = kr & 31, ntile = c >> 4, nb = c & 15;
      int lane_b = ((kk >> 4) << 4);
      int eb = ((kk & 15) & ~1) | (kk & 1);  // 2*((kk&15)>>1) + (kk&1)
      __bf16* dst = &sxB[((((ks << 2) + ntile) << 5) + lane_b + nb) * 16 + eb];
      dst[0]  = (__bf16)vv.x;
      dst[16] = (__bf16)vv.y;
      dst[32] = (__bf16)vv.z;
      dst[48] = (__bf16)vv.w;
    }
    __syncthreads();

#pragma unroll
    for (int ks = 0; ks < 8; ++ks) {
      // A: 16x32 from e (heads on M, m<8 valid)
      v16bf a;
      int m = lane & 15, halfk = lane >> 4;
#pragma unroll
      for (int v = 0; v < 8; ++v) {
        int k = ks * 32 + halfk * 16 + (v << 1);
        a[2 * v]     = (__bf16)((m < H_) ? se[(k << 3) | m] : 0.f);
        a[2 * v + 1] = (__bf16)((m < H_) ? se[((k + 1) << 3) | m] : 0.f);
      }
      v16bf bb = load_tile16(sxB, (ks << 2) + wave, lane);
      acc = __builtin_amdgcn_wmma_f32_16x16x32_bf16(
          false, a, false, bb, (short)0, acc, false, false);
    }
  }
  int n = lane & 15, half = lane >> 4;
#pragma unroll
  for (int r = 0; r < 8; ++r) {
    int m = r + 8 * half;
    if (m < H_)
      out[b * (H_ * F_) + m * F_ + f0 + wave * 16 + n] = acc[r] / denom[b * H_ + m];
  }
}

// ---------------- launch ----------------
extern "C" void kernel_launch(void* const* d_in, const int* in_sizes, int n_in,
                              void* d_out, int out_size, void* d_ws, size_t ws_size,
                              hipStream_t stream) {
  const float* x     = (const float*)d_in[0];
  const float* gum   = (const float*)d_in[1];
  const float* Wi    = (const float*)d_in[2];
  const float* b_in  = (const float*)d_in[3];
  const float* Wb    = (const float*)d_in[4];
  const float* b_blk = (const float*)d_in[5];
  const float* gamma = (const float*)d_in[6];
  const float* beta  = (const float*)d_in[7];
  const float* Wf    = (const float*)d_in[8];
  const float* b_fc  = (const float*)d_in[9];
  float* out = (float*)d_out;

  char* ws = (char*)d_ws;
  __bf16* Wi_sw = (__bf16*)(ws + 0);                 //  65536 B (64 swizzled tiles)
  __bf16* Wb_sw = (__bf16*)(ws + 65536);             //  32768 B (32 tiles)
  __bf16* Wf_sw = (__bf16*)(ws + 98304);             //   4096 B (4 tiles, N padded)
  float*  stats = (float*)(ws + 102400);             //   2048 B: sum[128], ss[128], denom[256]
  float*  chan_sum = stats;
  float*  chan_ss  = stats + 128;
  float*  denom    = stats + 256;
  float*  scale = (float*)(ws + 104448);             //    512 B
  float*  shift = scale + 128;                       //    512 B
  float*  mask  = (float*)(ws + 105472);             // 524288 B
  float*  e_ws  = (float*)(ws + 629760);             // 4194304 B
  __bf16* h_bf  = (__bf16*)(ws + 4824064);           // 33554432 B
  float*  z_ws  = (float*)(ws + 38378496);           // 67108864 B -> total ~100.6 MB

  k0_prep<<<64, 256, 0, stream>>>(Wi, Wb, Wf, Wi_sw, Wb_sw, Wf_sw, stats);
  k1_gemm12<<<N_ROWS / 64, 128, 0, stream>>>(x, Wi_sw, b_in, Wb_sw, b_blk,
                                             h_bf, z_ws, mask, chan_sum, chan_ss);
  k2_stats<<<1, 128, 0, stream>>>(chan_sum, chan_ss, gamma, beta, scale, shift);
  k3_fc_exp<<<N_ROWS / 64, 128, 0, stream>>>(z_ws, h_bf, Wf_sw, b_fc, gum, mask,
                                             scale, shift, e_ws, denom);
  k4_pool<<<dim3(B_, 4), 128, 0, stream>>>(x, e_ws, denom, out);
}
